// Aux_Free_Moe_47949014892923
// MI455X (gfx1250) — compile-verified, matrix operand order checked by reading, not run on hardware
//
#include <hip/hip_runtime.h>

// ---------------------------------------------------------------------------
// MoE (E=16, K=2) forward for MI455X (gfx1250, wave32, WMMA + TDM).
// Compute-bound (~687 GFLOP vs ~1GB traffic): bf16 WMMA GEMMs, TDM async
// tile staging into double-buffered LDS, 64x64 wave tiles (16 wmma / K-step).
// ---------------------------------------------------------------------------

#define BDIM 16384
#define FDIM 1024
#define HDIM 4096
#define ODIM 1024
#define EDIM 16
#define KTOP 2
#define CAP  2560   // ceil(1.25 * B * K / E)

typedef __attribute__((ext_vector_type(16))) __bf16 v16bf;
typedef __attribute__((ext_vector_type(8)))  float  v8f;

__device__ __forceinline__ unsigned short f2bf(float f) {
    unsigned u = __builtin_bit_cast(unsigned, f);
    unsigned r = u + 0x7FFFu + ((u >> 16) & 1u);   // round-to-nearest-even
    return (unsigned short)(r >> 16);
}

// 16x32 bf16 fragment from row-major [rows][32] LDS tile, per ISA layout:
// lanes 0-15 hold K=0-7,16-23; lanes 16-31 hold K=8-15,24-31.
__device__ __forceinline__ v16bf load_frag(const unsigned short* lds, int baseRow, int lane) {
    int row = baseRow + (lane & 15);
    int k0  = (lane < 16) ? 0 : 8;
    union { uint4 u[2]; v16bf v; } t;
    t.u[0] = *(const uint4*)(lds + row * 32 + k0);
    t.u[1] = *(const uint4*)(lds + row * 32 + k0 + 16);
    return t.v;
}

// ---------------------------------------------------------------------------
// Tensor Data Mover staging (gfx1250). Builds a 2D D# per ISA 08_async_tensor:
// group0 = {count=1 | lds_addr | global_addr(57b) | type=2},
// group1 = {data_size=2B | tensor_dim0=LD | tensor_dim1=rows | tile 32 x rows
//           | stride0=LD}. Groups 2/3 zero (2D tensor).
// ---------------------------------------------------------------------------
#if __has_builtin(__builtin_amdgcn_tensor_load_to_lds)
#define HAVE_TDM 1
typedef __attribute__((ext_vector_type(4))) unsigned int v4u;
typedef __attribute__((ext_vector_type(8))) int          v8i;
typedef __attribute__((ext_vector_type(4))) int          v4i;

__device__ __forceinline__ unsigned lds_addr_of(const void* p) {
    return (unsigned)(unsigned long long)p;   // generic LDS addr: low 32 bits = LDS offset
}

__device__ __forceinline__ void tdm_load_2d(unsigned lds_off, const void* gptr,
                                            unsigned ld_elems, unsigned rows) {
    unsigned long long ga = (unsigned long long)gptr;
    v4u g0;
    g0[0] = 1u;                                                   // count=1, user desc
    g0[1] = lds_off;                                              // LDS byte address
    g0[2] = (unsigned)ga;                                         // global_addr[31:0]
    g0[3] = (unsigned)((ga >> 32) & 0x01FFFFFFu) | (2u << 30);    // addr[56:32] | type=2
    v8i g1;
    g1[0] = (int)(1u << 16);                                      // data_size=1 -> 2 bytes
    g1[1] = (int)((ld_elems & 0xFFFFu) << 16);                    // tensor_dim0[15:0]
    g1[2] = (int)((ld_elems >> 16) | ((rows & 0xFFFFu) << 16));   // td0 hi | td1 lo
    g1[3] = (int)((rows >> 16) | (32u << 16));                    // td1 hi | tile_dim0=32
    g1[4] = (int)(rows & 0xFFFFu);                                // tile_dim1=rows, tile_dim2=0
    g1[5] = (int)ld_elems;                                        // stride0[31:0]
    g1[6] = 0;                                                    // stride0 hi | stride1 lo
    g1[7] = 0;
#if __clang_major__ >= 23
    __builtin_amdgcn_tensor_load_to_lds(g0, g1, (v4i)0, (v4i)0, (v8i)0, 0);
#else
    __builtin_amdgcn_tensor_load_to_lds(g0, g1, (v4i)0, (v4i)0, 0);
#endif
}
#else
#define HAVE_TDM 0
#endif

// ---------------------------------------------------------------------------
// 0) zero the output accumulator
// ---------------------------------------------------------------------------
__global__ void zero_f4_kernel(float4* __restrict__ p, size_t n4) {
    size_t i = (size_t)blockIdx.x * blockDim.x + threadIdx.x;
    if (i < n4) p[i] = float4{0.f, 0.f, 0.f, 0.f};
}

// ---------------------------------------------------------------------------
// 1) gating scores = feats @ Wg^T + bg + expert_bias   [B,E]
// ---------------------------------------------------------------------------
__global__ __launch_bounds__(256)
void gate_kernel(const float* __restrict__ feats, const float* __restrict__ Wg,
                 const float* __restrict__ bg, const float* __restrict__ ebias,
                 float* __restrict__ scores) {
    int tid = threadIdx.x;
    int e   = tid & 15;
    int t   = blockIdx.x * 16 + (tid >> 4);
    const float4* f = (const float4*)(feats + (size_t)t * FDIM);
    const float4* w = (const float4*)(Wg + (size_t)e * FDIM);
    float s = 0.f;
#pragma unroll 4
    for (int i = 0; i < FDIM / 4; ++i) {
        float4 a = f[i], b = w[i];
        s += a.x * b.x + a.y * b.y + a.z * b.z + a.w * b.w;
    }
    scores[(size_t)t * EDIM + e] = s + bg[e] + ebias[e];
}

// ---------------------------------------------------------------------------
// 2) top-2 + softmax(T=1). Ties pick lower index (jax top_k semantics).
// ---------------------------------------------------------------------------
__global__ void topk_kernel(const float* __restrict__ scores,
                            int* __restrict__ ti, float* __restrict__ tw) {
    int t = blockIdx.x * blockDim.x + threadIdx.x;
    const float* s = scores + (size_t)t * EDIM;
    float s0 = -3.4e38f; int i0 = 0;
#pragma unroll
    for (int e = 0; e < EDIM; ++e) { float v = s[e]; if (v > s0) { s0 = v; i0 = e; } }
    float s1 = -3.4e38f; int i1 = 0;
#pragma unroll
    for (int e = 0; e < EDIM; ++e) { float v = s[e]; if (e != i0 && v > s1) { s1 = v; i1 = e; } }
    float e2 = __expf(s1 - s0);
    float z  = 1.f + e2;
    ti[t * 2 + 0] = i0;  ti[t * 2 + 1] = i1;
    tw[t * 2 + 0] = 1.f / z;  tw[t * 2 + 1] = e2 / z;
}

// ---------------------------------------------------------------------------
// 3) routing: per-expert capacity-limited cumsum over batch order.
// ---------------------------------------------------------------------------
__global__ __launch_bounds__(1024)
void route_kernel(const int* __restrict__ ti, const float* __restrict__ tw,
                  int* __restrict__ inv, float* __restrict__ wslot) {
    __shared__ int s[1024];
    int e = blockIdx.x, tid = threadIdx.x;
    int carry = 0;
    for (int c = 0; c < BDIM / 1024; ++c) {
        int b  = c * 1024 + tid;
        int a0 = ti[b * 2], a1 = ti[b * 2 + 1];
        int flag = (a0 == e) || (a1 == e);
        float w  = (a0 == e) ? tw[b * 2] : ((a1 == e) ? tw[b * 2 + 1] : 0.f);
        s[tid] = flag;
        __syncthreads();
        for (int off = 1; off < 1024; off <<= 1) {
            int v = s[tid];
            if (tid >= off) v += s[tid - off];
            __syncthreads();
            s[tid] = v;
            __syncthreads();
        }
        int pos = carry + s[tid] - flag;
        if (flag && pos < CAP) {
            inv[e * CAP + pos]   = b;
            wslot[e * CAP + pos] = w;
        }
        carry += s[1023];
        __syncthreads();
    }
    int count = carry < CAP ? carry : CAP;
    for (int p = count + tid; p < CAP; p += 1024) inv[e * CAP + p] = BDIM;  // sentinel
}

// ---------------------------------------------------------------------------
// 4) transpose + fp32->bf16: src[e][R][C] -> dst[e][C][R]  (K-major weights)
// ---------------------------------------------------------------------------
__global__ __launch_bounds__(256)
void transpose_bf16_kernel(const float* __restrict__ src, unsigned short* __restrict__ dst,
                           int R, int C) {
    __shared__ float tile[32][33];
    int e  = blockIdx.z;
    int c0 = blockIdx.x * 32, r0 = blockIdx.y * 32;
    int tx = threadIdx.x & 31, ty = threadIdx.x >> 5;
    const float* sp = src + (size_t)e * R * C;
    unsigned short* dp = dst + (size_t)e * C * R;
#pragma unroll
    for (int j = 0; j < 32; j += 8)
        tile[ty + j][tx] = sp[(size_t)(r0 + ty + j) * C + (c0 + tx)];
    __syncthreads();
#pragma unroll
    for (int j = 0; j < 32; j += 8)
        dp[(size_t)(c0 + ty + j) * R + (r0 + tx)] = f2bf(tile[tx][ty + j]);
}

// ---------------------------------------------------------------------------
// 5) gather routed tokens -> Abuf bf16 [E][CAP][F]; sentinel slot -> zeros
// ---------------------------------------------------------------------------
__global__ __launch_bounds__(256)
void gather_kernel(const float* __restrict__ feats, const int* __restrict__ inv,
                   unsigned short* __restrict__ Abuf) {
    int e = blockIdx.y, slot = blockIdx.x, tid = threadIdx.x;
    int token = inv[e * CAP + slot];
    unsigned short* dst = Abuf + ((size_t)e * CAP + slot) * FDIM + tid * 4;
    if (token >= BDIM) {
        dst[0] = 0; dst[1] = 0; dst[2] = 0; dst[3] = 0;
    } else {
        float4 v = *(const float4*)(feats + (size_t)token * FDIM + tid * 4);
        dst[0] = f2bf(v.x); dst[1] = f2bf(v.y); dst[2] = f2bf(v.z); dst[3] = f2bf(v.w);
    }
}

// ---------------------------------------------------------------------------
// GEMM core: block tile 256x128, BK=32, 8 waves, wave tile 64x64 -> acc[4][4],
// 16 v_wmma_f32_16x16x32_bf16 per K-step. Double-buffered LDS:
//   As[2]: 256x32 bf16 (16KB each), Bs[2]: 128x32 bf16 (8KB each) = 48KB.
// A: [M][K] bf16, B(T): [N][K] bf16, both leading-dim LD.
// ---------------------------------------------------------------------------
#define GEMM_PROLOGUE()                                                         \
    __shared__ __align__(16) unsigned short As[2][256 * 32];                    \
    __shared__ __align__(16) unsigned short Bs[2][128 * 32];                    \
    const int tid   = threadIdx.x;                                              \
    const int lane  = tid & 31;                                                 \
    const int wid   = tid >> 5;                                                 \
    const int waveM = wid & 3;                                                  \
    const int waveN = wid >> 2;                                                 \
    v8f acc[4][4] = {};

#define GEMM_COMPUTE_STEP(Asc, Bsc)                                             \
    {                                                                           \
        v16bf af[4], bfr[4];                                                    \
        _Pragma("unroll")                                                       \
        for (int mi = 0; mi < 4; ++mi) af[mi] = load_frag((Asc), waveM * 64 + mi * 16, lane); \
        _Pragma("unroll")                                                       \
        for (int ni = 0; ni < 4; ++ni) bfr[ni] = load_frag((Bsc), waveN * 64 + ni * 16, lane); \
        _Pragma("unroll")                                                       \
        for (int mi = 0; mi < 4; ++mi)                                          \
            _Pragma("unroll")                                                   \
            for (int ni = 0; ni < 4; ++ni)                                      \
                acc[mi][ni] = __builtin_amdgcn_wmma_f32_16x16x32_bf16(          \
                    false, af[mi], false, bfr[ni], (short)0, acc[mi][ni], false, false); \
    }

#if HAVE_TDM
// One wave drives the TDM; s_wait_tensorcnt 2 retires the oldest (A,B) pair
// while the next pair streams into the other LDS buffer.
#define GEMM_KLOOP(Ap, Bp, LD, KSTEPS)                                          \
    {                                                                           \
        unsigned aOff[2] = { lds_addr_of(&As[0][0]), lds_addr_of(&As[1][0]) };  \
        unsigned bOff[2] = { lds_addr_of(&Bs[0][0]), lds_addr_of(&Bs[1][0]) };  \
        if (wid == 0) {                                                         \
            tdm_load_2d(aOff[0], (Ap), (LD), 256);                              \
            tdm_load_2d(bOff[0], (Bp), (LD), 128);                              \
        }                                                                       \
        for (int kt = 0; kt < (KSTEPS); ++kt) {                                 \
            int cur = kt & 1;                                                   \
            if (wid == 0) {                                                     \
                if (kt + 1 < (KSTEPS)) {                                        \
                    tdm_load_2d(aOff[cur ^ 1], (Ap) + (kt + 1) * 32, (LD), 256);\
                    tdm_load_2d(bOff[cur ^ 1], (Bp) + (kt + 1) * 32, (LD), 128);\
                    __builtin_amdgcn_s_wait_tensorcnt(2);                       \
                } else {                                                        \
                    __builtin_amdgcn_s_wait_tensorcnt(0);                       \
                }                                                               \
            }                                                                   \
            __syncthreads();                                                    \
            GEMM_COMPUTE_STEP(&As[cur][0], &Bs[cur][0])                         \
            __syncthreads();                                                    \
        }                                                                       \
    }
#else
// Fallback: per-thread double-buffered copy; global loads for tile kt+1 issue
// before the compute of tile kt, LDS stores land after it.
#define GEMM_KLOOP(Ap, Bp, LD, KSTEPS)                                          \
    {                                                                           \
        uint4 ra[4], rb[2];                                                     \
        const int bRow = tid >> 1, bCol = (tid & 1) * 16;                       \
        _Pragma("unroll")                                                       \
        for (int j = 0; j < 4; ++j) ra[j] = *(const uint4*)((Ap) + (size_t)tid * (LD) + j * 8); \
        _Pragma("unroll")                                                       \
        for (int j = 0; j < 2; ++j) rb[j] = *(const uint4*)((Bp) + (size_t)bRow * (LD) + bCol + j * 8); \
        _Pragma("unroll")                                                       \
        for (int j = 0; j < 4; ++j) *(uint4*)(&As[0][0] + tid * 32 + j * 8) = ra[j]; \
        _Pragma("unroll")                                                       \
        for (int j = 0; j < 2; ++j) *(uint4*)(&Bs[0][0] + bRow * 32 + bCol + j * 8) = rb[j]; \
        for (int kt = 0; kt < (KSTEPS); ++kt) {                                 \
            int cur = kt & 1;                                                   \
            if (kt + 1 < (KSTEPS)) {                                            \
                const unsigned short* ga = (Ap) + (size_t)tid * (LD) + (kt + 1) * 32; \
                const unsigned short* gb = (Bp) + (size_t)bRow * (LD) + (kt + 1) * 32 + bCol; \
                _Pragma("unroll")                                               \
                for (int j = 0; j < 4; ++j) ra[j] = *(const uint4*)(ga + j * 8);\
                _Pragma("unroll")                                               \
                for (int j = 0; j < 2; ++j) rb[j] = *(const uint4*)(gb + j * 8);\
                __builtin_prefetch(ga + 32, 0, 1);                              \
            }                                                                   \
            __syncthreads();                                                    \
            GEMM_COMPUTE_STEP(&As[cur][0], &Bs[cur][0])                         \
            if (kt + 1 < (KSTEPS)) {                                            \
                _Pragma("unroll")                                               \
                for (int j = 0; j < 4; ++j) *(uint4*)(&As[cur ^ 1][0] + tid * 32 + j * 8) = ra[j]; \
                _Pragma("unroll")                                               \
                for (int j = 0; j < 2; ++j) *(uint4*)(&Bs[cur ^ 1][0] + bRow * 32 + bCol + j * 8) = rb[j]; \
            }                                                                   \
            __syncthreads();                                                    \
        }                                                                       \
    }
#endif

// GEMM1: h = relu(Abuf @ W1T^T + b1) -> bf16.  M=CAP, N=HDIM, K=FDIM.
__global__ __launch_bounds__(256)
void gemm1_kernel(const unsigned short* __restrict__ Abuf,   // [E][CAP][F]
                  const unsigned short* __restrict__ W1T,    // [E][H][F]
                  const float* __restrict__ b1,              // [E][H]
                  unsigned short* __restrict__ Hbuf) {       // [E][CAP][H]
    const int e  = blockIdx.y;
    const int nt = blockIdx.x % (HDIM / 128);
    const int mt = blockIdx.x / (HDIM / 128);
    const unsigned short* Ap = Abuf + (size_t)e * CAP * FDIM + (size_t)mt * 256 * FDIM;
    const unsigned short* Bp = W1T  + (size_t)e * HDIM * FDIM + (size_t)nt * 128 * FDIM;
    GEMM_PROLOGUE()
    GEMM_KLOOP(Ap, Bp, FDIM, FDIM / 32)
    const int l15 = lane & 15;
    const int rHalf = (lane < 16) ? 0 : 8;
#pragma unroll
    for (int mi = 0; mi < 4; ++mi)
#pragma unroll
        for (int ni = 0; ni < 4; ++ni) {
            int ncol = nt * 128 + waveN * 64 + ni * 16 + l15;
            float bias = b1[e * HDIM + ncol];
#pragma unroll
            for (int r = 0; r < 8; ++r) {
                int m = mt * 256 + waveM * 64 + mi * 16 + rHalf + r;
                float v = acc[mi][ni][r] + bias;
                v = v > 0.f ? v : 0.f;
                Hbuf[(size_t)e * CAP * HDIM + (size_t)m * HDIM + ncol] = f2bf(v);
            }
        }
}

// GEMM2: y = Hbuf @ W2T^T + b2, scale by gate weight, atomic scatter to out.
__global__ __launch_bounds__(256)
void gemm2_kernel(const unsigned short* __restrict__ Hbuf,   // [E][CAP][H]
                  const unsigned short* __restrict__ W2T,    // [E][O][H]
                  const float* __restrict__ b2,              // [E][O]
                  const int* __restrict__ inv,               // [E][CAP]
                  const float* __restrict__ wslot,           // [E][CAP]
                  float* __restrict__ out) {                 // [B][O]
    const int e  = blockIdx.y;
    const int nt = blockIdx.x % (ODIM / 128);
    const int mt = blockIdx.x / (ODIM / 128);
    const unsigned short* Ap = Hbuf + (size_t)e * CAP * HDIM + (size_t)mt * 256 * HDIM;
    const unsigned short* Bp = W2T  + (size_t)e * ODIM * HDIM + (size_t)nt * 128 * HDIM;
    GEMM_PROLOGUE()
    GEMM_KLOOP(Ap, Bp, HDIM, HDIM / 32)
    const int l15 = lane & 15;
    const int rHalf = (lane < 16) ? 0 : 8;
#pragma unroll
    for (int mi = 0; mi < 4; ++mi) {
        int tok[8]; float wgt[8];
#pragma unroll
        for (int r = 0; r < 8; ++r) {
            int m = mt * 256 + waveM * 64 + mi * 16 + rHalf + r;
            tok[r] = inv[e * CAP + m];
            wgt[r] = wslot[e * CAP + m];
        }
#pragma unroll
        for (int ni = 0; ni < 4; ++ni) {
            int ncol = nt * 128 + waveN * 64 + ni * 16 + l15;
            float bias = b2[e * ODIM + ncol];
#pragma unroll
            for (int r = 0; r < 8; ++r) {
                if (tok[r] < BDIM) {
                    float v = (acc[mi][ni][r] + bias) * wgt[r];
                    atomicAdd(out + (size_t)tok[r] * ODIM + ncol, v);
                }
            }
        }
    }
}

// ---------------------------------------------------------------------------
extern "C" void kernel_launch(void* const* d_in, const int* in_sizes, int n_in,
                              void* d_out, int out_size, void* d_ws, size_t ws_size,
                              hipStream_t stream) {
    const float* feats = (const float*)d_in[0];
    const float* Wg    = (const float*)d_in[1];
    const float* bg    = (const float*)d_in[2];
    const float* W1    = (const float*)d_in[3];
    const float* b1    = (const float*)d_in[4];
    const float* W2    = (const float*)d_in[5];
    const float* b2    = (const float*)d_in[6];
    const float* ebias = (const float*)d_in[7];
    float* out = (float*)d_out;

    char* ws = (char*)d_ws;
    size_t off = 0;
    auto alloc = [&](size_t bytes) -> void* {
        void* p = ws + off;
        off += (bytes + 255) & ~(size_t)255;
        return p;
    };
    float*          scores = (float*)alloc((size_t)BDIM * EDIM * 4);
    int*            tpk_i  = (int*)  alloc((size_t)BDIM * KTOP * 4);
    float*          tpk_w  = (float*)alloc((size_t)BDIM * KTOP * 4);
    int*            inv    = (int*)  alloc((size_t)EDIM * CAP * 4);
    float*          wslot  = (float*)alloc((size_t)EDIM * CAP * 4);
    unsigned short* W1T    = (unsigned short*)alloc((size_t)EDIM * HDIM * FDIM * 2);
    unsigned short* W2T    = (unsigned short*)alloc((size_t)EDIM * ODIM * HDIM * 2);
    unsigned short* Abuf   = (unsigned short*)alloc((size_t)EDIM * CAP * FDIM * 2);
    unsigned short* Hbuf   = (unsigned short*)alloc((size_t)EDIM * CAP * HDIM * 2);
    (void)ws_size; (void)in_sizes; (void)n_in; (void)out_size;

    size_t n4 = (size_t)BDIM * ODIM / 4;
    zero_f4_kernel<<<dim3((unsigned)((n4 + 255) / 256)), dim3(256), 0, stream>>>((float4*)out, n4);
    gate_kernel<<<dim3(BDIM / 16), dim3(256), 0, stream>>>(feats, Wg, bg, ebias, scores);
    topk_kernel<<<dim3(BDIM / 256), dim3(256), 0, stream>>>(scores, tpk_i, tpk_w);
    route_kernel<<<dim3(EDIM), dim3(1024), 0, stream>>>(tpk_i, tpk_w, inv, wslot);
    transpose_bf16_kernel<<<dim3(HDIM / 32, FDIM / 32, EDIM), dim3(256), 0, stream>>>(W1, W1T, FDIM, HDIM);
    transpose_bf16_kernel<<<dim3(ODIM / 32, HDIM / 32, EDIM), dim3(256), 0, stream>>>(W2, W2T, HDIM, ODIM);
    gather_kernel<<<dim3(CAP, EDIM), dim3(256), 0, stream>>>(feats, inv, Abuf);
    gemm1_kernel<<<dim3((CAP / 256) * (HDIM / 128), EDIM), dim3(256), 0, stream>>>(Abuf, W1T, b1, Hbuf);
    gemm2_kernel<<<dim3((CAP / 256) * (ODIM / 128), EDIM), dim3(256), 0, stream>>>(Hbuf, W2T, b2, inv, wslot, out);
}
